// TransformerEncoderLayer_33011118637460
// MI455X (gfx1250) — matrix operand order repaired and checked by reading, not compiled, not run on hardware
//
#include <hip/hip_runtime.h>
#include <hip/hip_bf16.h>

// ---------------------------------------------------------------------------
// Types for CDNA5 WMMA (gfx1250, wave32)
// ---------------------------------------------------------------------------
typedef __attribute__((ext_vector_type(16))) __bf16 bf16x16;
typedef __attribute__((ext_vector_type(8)))  float  f32x8;
typedef __attribute__((ext_vector_type(4)))  int    v4i;

#define DIMC     1024
#define NSEQ     1024
#define NBATCH   4
#define NHEADS   16
#define HDIM     64
#define ATT_SCALE 0.125f   // HEAD_DIM^-0.5

#if __has_builtin(__builtin_amdgcn_global_load_async_to_lds_b128)
#define ASYNC_LDS 1
#endif

__device__ __forceinline__ unsigned short f2bf(float f) {
  unsigned u = __float_as_uint(f);
  u += 0x7FFFu + ((u >> 16) & 1u);           // round-to-nearest-even
  return (unsigned short)(u >> 16);
}

// 16B global -> LDS copy: async (ASYNCcnt) on CDNA5 toolchains that expose it.
// Builtin signature (from clang diagnostic): (v4i AS(1)*, v4i AS(3)*, imm, imm)
__device__ __forceinline__ void cp_g2s_b128(const unsigned short* g,
                                            unsigned short* s) {
#ifdef ASYNC_LDS
  __builtin_amdgcn_global_load_async_to_lds_b128(
      (__attribute__((address_space(1))) v4i*)(unsigned short*)g,
      (__attribute__((address_space(3))) v4i*)s, 0, 0);
#else
  *(uint4*)s = *(const uint4*)g;
#endif
}

__device__ __forceinline__ void cp_wait() {
#ifdef ASYNC_LDS
#if __has_builtin(__builtin_amdgcn_s_wait_asynccnt)
  __builtin_amdgcn_s_wait_asynccnt(0);
#else
  asm volatile("s_wait_asynccnt 0x0" ::: "memory");
#endif
#endif
}

__device__ __forceinline__ f32x8 wmma_bf16(bf16x16 a, bf16x16 b, f32x8 c) {
  // D = A(16x32 bf16) * B(32x16 bf16) + C(16x16 f32)
  return __builtin_amdgcn_wmma_f32_16x16x32_bf16(
      /*neg_a=*/false, a, /*neg_b=*/false, b,
      /*c_mod=*/(short)0, c, /*reuse_a=*/false, /*reuse_b=*/false);
}

union FragU { bf16x16 v; unsigned u[8]; };

// A-fragment (16x32, bf16) from row-major ushort LDS tile, row stride ldp.
// Lane L: m=L&15, k-base=8*(L>>4); dwords: k = kb+2d (d<4), 16+kb+2(d-4) (d>=4)
__device__ __forceinline__ bf16x16 load_afrag(const unsigned short* base, int ldp,
                                              int koff, int lane) {
  const int m = lane & 15, half = lane >> 4;
  const unsigned* rp = (const unsigned*)(base + m * ldp + koff);
  const int kb = half * 4;                    // dword offset of k-base
  FragU f;
#pragma unroll
  for (int d = 0; d < 4; ++d) f.u[d]     = rp[kb + d];
#pragma unroll
  for (int d = 0; d < 4; ++d) f.u[4 + d] = rp[8 + kb + d];
  return f.v;
}

// B-fragment (32x16, bf16) from LDS stored as [n][k] (row stride ldp).
// Lane L: n=L&15, k-base=8*(L>>4); dwords: k = kb+2d (d<4), 16+kb+2(d-4) (d>=4)
__device__ __forceinline__ bf16x16 load_bfrag(const unsigned short* base, int ldp,
                                              int koff, int lane) {
  const int n = lane & 15, half = lane >> 4;
  const unsigned* rp = (const unsigned*)(base + n * ldp + koff);
  FragU f;
#pragma unroll
  for (int d = 0; d < 8; ++d) f.u[d] = rp[(d >> 2) * 8 + half * 4 + (d & 3)];
  return f.v;
}

// ---------------------------------------------------------------------------
// Weight convert + transpose: W[K][N] fp32 -> Wt[N][K] bf16 (32x32 LDS tiles)
// ---------------------------------------------------------------------------
__global__ __launch_bounds__(256)
void wtrans_kernel(const float* __restrict__ W, unsigned short* __restrict__ Wt,
                   int K, int N) {
  __shared__ unsigned short t[32][33];
  const int n0 = blockIdx.x * 32, k0 = blockIdx.y * 32;
  const int x = threadIdx.x & 31, y = threadIdx.x >> 5;   // x:0..31, y:0..7
#pragma unroll
  for (int j = 0; j < 4; ++j)
    t[y + j * 8][x] = f2bf(W[(size_t)(k0 + y + j * 8) * N + n0 + x]);
  __syncthreads();
#pragma unroll
  for (int j = 0; j < 4; ++j)
    Wt[(size_t)(n0 + y + j * 8) * K + k0 + x] = t[x][y + j * 8];
}

// ---------------------------------------------------------------------------
// V transpose: KV[b,n,1024 + h*64 + d] -> Vt[(b*16+h)*64 + d][n]  (bf16)
// ---------------------------------------------------------------------------
__global__ __launch_bounds__(256)
void vtrans_kernel(const unsigned short* __restrict__ KVb,
                   unsigned short* __restrict__ Vt) {
  __shared__ unsigned short t[32][33];
  const int n0 = blockIdx.x * 32, d0 = blockIdx.y * 32, bh = blockIdx.z;
  const int b = bh >> 4, h = bh & 15;
  const int x = threadIdx.x & 31, y = threadIdx.x >> 5;
#pragma unroll
  for (int j = 0; j < 4; ++j)
    t[y + j * 8][x] = KVb[((size_t)b * NSEQ + n0 + y + j * 8) * (2 * DIMC) +
                          DIMC + h * HDIM + d0 + x];
  __syncthreads();
#pragma unroll
  for (int j = 0; j < 4; ++j)
    Vt[((size_t)bh * HDIM + d0 + y + j * 8) * NSEQ + n0 + x] = t[x][y + j * 8];
}

// ---------------------------------------------------------------------------
// LayerNorm: fp32 [row,1024] -> bf16, one block per row
// ---------------------------------------------------------------------------
__global__ __launch_bounds__(256)
void ln_kernel(const float* __restrict__ x, const float* __restrict__ g,
               const float* __restrict__ bb, unsigned short* __restrict__ out) {
  __shared__ float r1[256];
  __shared__ float r2[256];
  const int row = blockIdx.x, t = threadIdx.x;
  float4 v = ((const float4*)(x + (size_t)row * DIMC))[t];
  r1[t] = v.x + v.y + v.z + v.w;
  r2[t] = v.x * v.x + v.y * v.y + v.z * v.z + v.w * v.w;
  __syncthreads();
#pragma unroll
  for (int st = 128; st > 0; st >>= 1) {
    if (t < st) { r1[t] += r1[t + st]; r2[t] += r2[t + st]; }
    __syncthreads();
  }
  const float mean = r1[0] * (1.0f / DIMC);
  const float var  = r2[0] * (1.0f / DIMC) - mean * mean;
  const float rstd = rsqrtf(var + 1e-5f);
  float4 gv = ((const float4*)g)[t];
  float4 bv = ((const float4*)bb)[t];
  unsigned short* op = out + (size_t)row * DIMC + t * 4;
  op[0] = f2bf((v.x - mean) * rstd * gv.x + bv.x);
  op[1] = f2bf((v.y - mean) * rstd * gv.y + bv.y);
  op[2] = f2bf((v.z - mean) * rstd * gv.z + bv.z);
  op[3] = f2bf((v.w - mean) * rstd * gv.w + bv.w);
}

// ---------------------------------------------------------------------------
// Tiled WMMA GEMM: C[M,N] = A[M,K](bf16) x Bt[N,K](bf16, pre-transposed)
// MODE 0: out bf16
// MODE 1: out fp32 = acc + bias[col] + res[row,col]
// MODE 2: out bf16 = gelu_exact(acc + bias[col])
// 256 thr (8 waves), tile 128x128, BK=32, double-buffered async LDS staging.
// ---------------------------------------------------------------------------
#define BM 128
#define BN 128
#define BK 32
#define LDP 40   // padded LDS row (elements): conflict-free fragment reads

template <int MODE>
__global__ __launch_bounds__(256)
void gemm_bf16(const unsigned short* __restrict__ A,
               const unsigned short* __restrict__ Bt,
               void* __restrict__ Out,
               const float* __restrict__ bias,
               const float* __restrict__ res,
               int M, int N, int K) {
  __shared__ unsigned short a_s[2][BM * LDP];   // [m][k]
  __shared__ unsigned short b_s[2][BN * LDP];   // [n][k]

  const int tid  = threadIdx.x;
  const int lane = tid & 31;
  const int w    = tid >> 5;
  const int wm   = w >> 2;                   // 0..1 -> 64-row band
  const int wn   = w & 3;                    // 0..3 -> 32-col band
  const int row0 = blockIdx.y * BM;
  const int col0 = blockIdx.x * BN;

  f32x8 acc[4][2];
  const f32x8 zero = {0.f, 0.f, 0.f, 0.f, 0.f, 0.f, 0.f, 0.f};
#pragma unroll
  for (int i = 0; i < 4; ++i)
#pragma unroll
    for (int j = 0; j < 2; ++j) acc[i][j] = zero;

  auto stage = [&](int buf, int kt) {
#pragma unroll
    for (int it = 0; it < 2; ++it) {
      int idx = tid + it * 256;
      int r = idx >> 2, c = (idx & 3) * 8;        // r:0..127, c:{0,8,16,24}
      cp_g2s_b128(A  + (size_t)(row0 + r) * K + kt + c, &a_s[buf][r * LDP + c]);
      cp_g2s_b128(Bt + (size_t)(col0 + r) * K + kt + c, &b_s[buf][r * LDP + c]);
    }
  };

  const int nk = K / BK;
  stage(0, 0);
  cp_wait();
  __syncthreads();

  for (int t = 0; t < nk; ++t) {
    const int cur = t & 1;
    if (t + 1 < nk) stage(cur ^ 1, (t + 1) * BK);   // overlap with compute

    bf16x16 af[4], bfm[2];
#pragma unroll
    for (int i = 0; i < 4; ++i)
      af[i] = load_afrag(&a_s[cur][(wm * 64 + i * 16) * LDP], LDP, 0, lane);
#pragma unroll
    for (int j = 0; j < 2; ++j)
      bfm[j] = load_bfrag(&b_s[cur][(wn * 32 + j * 16) * LDP], LDP, 0, lane);
#pragma unroll
    for (int i = 0; i < 4; ++i)
#pragma unroll
      for (int j = 0; j < 2; ++j) acc[i][j] = wmma_bf16(af[i], bfm[j], acc[i][j]);

    cp_wait();
    __syncthreads();
  }

  // --- epilogue: C layout lane n=L&15, VGPR v -> row v + 8*(L>=16) ---
  const int half = lane >> 4, n = lane & 15;
#pragma unroll
  for (int i = 0; i < 4; ++i) {
#pragma unroll
    for (int j = 0; j < 2; ++j) {
      const int col = col0 + wn * 32 + j * 16 + n;
#pragma unroll
      for (int v = 0; v < 8; ++v) {
        const int r = row0 + wm * 64 + i * 16 + v + 8 * half;
        const size_t o = (size_t)r * N + col;
        float x = acc[i][j][v];
        if (MODE == 0) {
          ((unsigned short*)Out)[o] = f2bf(x);
        } else if (MODE == 1) {
          ((float*)Out)[o] = x + bias[col] + res[o];
        } else {
          float tt = x + bias[col];
          float gl = 0.5f * tt * (1.0f + erff(tt * 0.70710678118654752f));
          ((unsigned short*)Out)[o] = f2bf(gl);
        }
      }
    }
  }
}

// ---------------------------------------------------------------------------
// Flash attention: block = (qtile, head, batch), 128 thr (4 waves x 16 rows)
// Q[b,n,h*64+d] bf16; K from KV row-major; V from Vt[(b*16+h)*64+d][n]
// out X[b,n,h*64+d] bf16 = (softmax(QK^T * s) V) * s
// ---------------------------------------------------------------------------
__global__ __launch_bounds__(128)
void attn_kernel(const unsigned short* __restrict__ Qb,
                 const unsigned short* __restrict__ KVb,
                 const unsigned short* __restrict__ Vt,
                 unsigned short* __restrict__ Xb) {
  __shared__ unsigned short qs[64 * 72];        // [qrow][dim]
  __shared__ unsigned short ks[64 * 72];        // [key][dim]
  __shared__ unsigned short vs[64 * 72];        // [dim][key]
  __shared__ unsigned short ps[4 * 16 * 72];    // per-wave P tile [m][key]

  const int tid = threadIdx.x, lane = tid & 31, w = tid >> 5;
  const int qt = blockIdx.x, h = blockIdx.y, b = blockIdx.z;
  const size_t qrow0 = (size_t)b * NSEQ + qt * 64;
  const int qcol = h * HDIM;
  const int half = lane >> 4, ln16 = lane & 15;
  const unsigned short* Vth = Vt + (size_t)(b * NHEADS + h) * HDIM * NSEQ;

  // stage Q tile 64x64
#pragma unroll
  for (int it = 0; it < 4; ++it) {
    int idx = tid + it * 128;
    int r = idx >> 3, c = (idx & 7) * 8;
    cp_g2s_b128(Qb + (qrow0 + r) * DIMC + qcol + c, &qs[r * 72 + c]);
  }
  cp_wait();
  __syncthreads();

  bf16x16 aq0 = load_afrag(qs + (w * 16) * 72, 72, 0, lane);
  bf16x16 aq1 = load_afrag(qs + (w * 16) * 72, 72, 32, lane);

  const f32x8 zero = {0.f, 0.f, 0.f, 0.f, 0.f, 0.f, 0.f, 0.f};
  f32x8 o[4];
#pragma unroll
  for (int i = 0; i < 4; ++i) o[i] = zero;
  float mst[8], lst[8];
#pragma unroll
  for (int v = 0; v < 8; ++v) { mst[v] = -1e30f; lst[v] = 0.f; }

  unsigned short* psw = ps + w * 16 * 72;

  for (int kt = 0; kt < NSEQ; kt += 64) {
    __syncthreads();
    // stage K (row-major from KV) and V (row-major from Vt)
#pragma unroll
    for (int it = 0; it < 4; ++it) {
      int idx = tid + it * 128;
      int r = idx >> 3, c = (idx & 7) * 8;
      cp_g2s_b128(KVb + ((size_t)b * NSEQ + kt + r) * (2 * DIMC) + qcol + c,
                  &ks[r * 72 + c]);
      cp_g2s_b128(Vth + (size_t)r * NSEQ + kt + c, &vs[r * 72 + c]);
    }
    cp_wait();
    __syncthreads();

    // S = Q K^T  (16 q-rows x 64 keys per wave)
    float sv[4][8];
#pragma unroll
    for (int nt = 0; nt < 4; ++nt) {
      f32x8 s = zero;
      s = wmma_bf16(aq0, load_bfrag(ks + nt * 16 * 72, 72, 0, lane), s);
      s = wmma_bf16(aq1, load_bfrag(ks + nt * 16 * 72, 72, 32, lane), s);
#pragma unroll
      for (int v = 0; v < 8; ++v) sv[nt][v] = s[v] * ATT_SCALE;
    }

    // online softmax (row = v + 8*half; 64 cols = 16 lanes x 4 tiles)
#pragma unroll
    for (int v = 0; v < 8; ++v) {
      float rm = fmaxf(fmaxf(sv[0][v], sv[1][v]), fmaxf(sv[2][v], sv[3][v]));
      rm = fmaxf(rm, __shfl_xor(rm, 1, 32));
      rm = fmaxf(rm, __shfl_xor(rm, 2, 32));
      rm = fmaxf(rm, __shfl_xor(rm, 4, 32));
      rm = fmaxf(rm, __shfl_xor(rm, 8, 32));
      float mnew = fmaxf(mst[v], rm);
      float corr = __expf(mst[v] - mnew);
      mst[v] = mnew;
      float rs = 0.f;
#pragma unroll
      for (int nt = 0; nt < 4; ++nt) {
        float p = __expf(sv[nt][v] - mnew);
        sv[nt][v] = p;
        rs += p;
      }
      rs += __shfl_xor(rs, 1, 32);
      rs += __shfl_xor(rs, 2, 32);
      rs += __shfl_xor(rs, 4, 32);
      rs += __shfl_xor(rs, 8, 32);
      lst[v] = lst[v] * corr + rs;
#pragma unroll
      for (int dt = 0; dt < 4; ++dt) o[dt][v] *= corr;
    }

    // P: C-layout regs -> per-wave LDS (bf16) -> A-layout frags
#pragma unroll
    for (int v = 0; v < 8; ++v) {
      int m = v + 8 * half;
#pragma unroll
      for (int nt = 0; nt < 4; ++nt)
        psw[m * 72 + nt * 16 + ln16] = f2bf(sv[nt][v]);
    }
    __syncthreads();
    bf16x16 ap0 = load_afrag(psw, 72, 0, lane);
    bf16x16 ap1 = load_afrag(psw, 72, 32, lane);

    // O += P V
#pragma unroll
    for (int dt = 0; dt < 4; ++dt) {
      o[dt] = wmma_bf16(ap0, load_bfrag(vs + dt * 16 * 72, 72, 0, lane), o[dt]);
      o[dt] = wmma_bf16(ap1, load_bfrag(vs + dt * 16 * 72, 72, 32, lane), o[dt]);
    }
  }

  // normalize (x 1/l) and apply the reference's second SCALE, write bf16
#pragma unroll
  for (int v = 0; v < 8; ++v) {
    float inv = ATT_SCALE / lst[v];
    size_t r = qrow0 + w * 16 + v + 8 * half;
#pragma unroll
    for (int dt = 0; dt < 4; ++dt)
      Xb[r * DIMC + qcol + dt * 16 + ln16] = f2bf(o[dt][v] * inv);
  }
}

// ---------------------------------------------------------------------------
// Host-side orchestration
// ---------------------------------------------------------------------------
extern "C" void kernel_launch(void* const* d_in, const int* in_sizes, int n_in,
                              void* d_out, int out_size, void* d_ws, size_t ws_size,
                              hipStream_t stream) {
  const float* src_q  = (const float*)d_in[0];
  const float* src_kv = (const float*)d_in[1];
  const float* gq     = (const float*)d_in[2];
  const float* bq     = (const float*)d_in[3];
  const float* gkv    = (const float*)d_in[4];
  const float* bkv    = (const float*)d_in[5];
  const float* Wq     = (const float*)d_in[6];
  const float* Wkv    = (const float*)d_in[7];
  const float* Wproj  = (const float*)d_in[8];
  const float* bproj  = (const float*)d_in[9];
  const float* gn     = (const float*)d_in[10];
  const float* bn     = (const float*)d_in[11];
  const float* W1     = (const float*)d_in[12];
  const float* b1     = (const float*)d_in[13];
  const float* W2     = (const float*)d_in[14];
  const float* b2     = (const float*)d_in[15];
  float* out = (float*)d_out;

  char* ws = (char*)d_ws;
  // workspace layout (96 MB total)
  unsigned short* qn    = (unsigned short*)(ws);               //  8 MB [0,8)
  unsigned short* kvn   = (unsigned short*)(ws + (8u  << 20)); //  8 MB [8,16)
  unsigned short* wq_t  = (unsigned short*)(ws + (16u << 20)); //  2 MB
  unsigned short* wkv_t = (unsigned short*)(ws + (18u << 20)); //  4 MB
  unsigned short* wp_t  = (unsigned short*)(ws + (22u << 20)); //  2 MB
  unsigned short* w1_t  = (unsigned short*)(ws + (24u << 20)); //  8 MB
  unsigned short* w2_t  = (unsigned short*)(ws + (32u << 20)); //  8 MB
  unsigned short* Qb    = (unsigned short*)(ws + (40u << 20)); //  8 MB [40,48)
  unsigned short* KVb   = (unsigned short*)(ws + (48u << 20)); // 16 MB [48,64)
  unsigned short* Xb    = (unsigned short*)(ws + (64u << 20)); //  8 MB [64,72)
  float*          srcb  = (float*)        (ws + (72u << 20)); // 16 MB [72,88)
  unsigned short* Vtb   = (unsigned short*)(ws + (88u << 20)); //  8 MB [88,96)
  unsigned short* Hb    = Qb;        // reuse [40,72): Q/KV/X dead by MLP1
  unsigned short* srcn  = qn;        // reuse [0,8):   qn dead after Q GEMM

  // 1. weight fp32 -> bf16, transposed to [N][K]
  wtrans_kernel<<<dim3(1024 / 32, 1024 / 32), 256, 0, stream>>>(Wq,    wq_t,  1024, 1024);
  wtrans_kernel<<<dim3(2048 / 32, 1024 / 32), 256, 0, stream>>>(Wkv,   wkv_t, 1024, 2048);
  wtrans_kernel<<<dim3(1024 / 32, 1024 / 32), 256, 0, stream>>>(Wproj, wp_t,  1024, 1024);
  wtrans_kernel<<<dim3(4096 / 32, 1024 / 32), 256, 0, stream>>>(W1,    w1_t,  1024, 4096);
  wtrans_kernel<<<dim3(1024 / 32, 4096 / 32), 256, 0, stream>>>(W2,    w2_t,  4096, 1024);

  // 2. layernorms -> bf16 activations
  ln_kernel<<<4096, 256, 0, stream>>>(src_q,  gq,  bq,  qn);
  ln_kernel<<<4096, 256, 0, stream>>>(src_kv, gkv, bkv, kvn);

  // 3. Q = qn Wq ; KV = kvn Wkv
  gemm_bf16<0><<<dim3(1024 / BN, 4096 / BM), 256, 0, stream>>>(
      qn, wq_t, Qb, nullptr, nullptr, 4096, 1024, 1024);
  gemm_bf16<0><<<dim3(2048 / BN, 4096 / BM), 256, 0, stream>>>(
      kvn, wkv_t, KVb, nullptr, nullptr, 4096, 2048, 1024);

  // 4. V transpose for attention, then attention
  vtrans_kernel<<<dim3(NSEQ / 32, HDIM / 32, NBATCH * NHEADS), 256, 0, stream>>>(KVb, Vtb);
  attn_kernel<<<dim3(NSEQ / 64, NHEADS, NBATCH), 128, 0, stream>>>(Qb, KVb, Vtb, Xb);

  // 5. proj + bias + residual(src_q) -> srcb (fp32)
  gemm_bf16<1><<<dim3(1024 / BN, 4096 / BM), 256, 0, stream>>>(
      Xb, wp_t, srcb, bproj, src_q, 4096, 1024, 1024);

  // 6. LN -> srcn (bf16)
  ln_kernel<<<4096, 256, 0, stream>>>(srcb, gn, bn, srcn);

  // 7. MLP1: gelu(srcn W1 + b1) -> Hb (bf16)
  gemm_bf16<2><<<dim3(4096 / BN, 4096 / BM), 256, 0, stream>>>(
      srcn, w1_t, Hb, b1, nullptr, 4096, 4096, 1024);

  // 8. MLP2: Hb W2 + b2 + srcb -> out (fp32)
  gemm_bf16<1><<<dim3(1024 / BN, 4096 / BM), 256, 0, stream>>>(
      Hb, w2_t, out, b2, srcb, 4096, 1024, 4096);
}
